// CausalSelfAttention_45947560132787
// MI455X (gfx1250) — matrix operand order repaired and checked by reading, not compiled, not run on hardware
//
#include <hip/hip_runtime.h>
#include <hip/hip_bf16.h>

// ---------------------------------------------------------------------------
// CDNA5 (gfx1250) causal self-attention, bf16 WMMA pipeline.
//   B=512, T=256, C=384, 3C=1152.  All matmuls via v_wmma_f32_16x16x32_bf16.
//   Round 3: __builtin_amdgcn_sched_barrier(0) pins bulk fragment preloads so
//   the scheduler cannot re-serialize them into load->wait0->wmma chains.
// ---------------------------------------------------------------------------

typedef __attribute__((ext_vector_type(16))) __bf16 v16bf;
typedef __attribute__((ext_vector_type(8)))  __bf16 v8bf;
typedef __attribute__((ext_vector_type(4)))  __bf16 v4bf;
typedef __attribute__((ext_vector_type(8)))  float  v8f;
typedef __attribute__((ext_vector_type(4)))  float  v4f;

#define BATCH 512
#define SEQ   256
#define CH    384
#define CH3   1152

__device__ __forceinline__ __bf16 f2bf(float f) {
  // round-to-nearest-even float -> bf16 via integer math
  unsigned u = __builtin_bit_cast(unsigned, f);
  unsigned r = (u + 0x7FFFu + ((u >> 16) & 1u)) >> 16;
  unsigned short s = (unsigned short)r;
  return __builtin_bit_cast(__bf16, s);
}

// A fragment (16x32 bf16, M x K) from row-major src (ld elements per row).
// Lane L: m = L%16, half = L/16.  Two contiguous 16B runs per lane:
//   K = half*8 + [0..7]  and  K = 16 + half*8 + [0..7]
__device__ __forceinline__ v16bf load_frag_a(const __bf16* src, int ld) {
  const int lane = threadIdx.x & 31;
  const int m    = lane & 15;
  const int half = lane >> 4;
  const __bf16* p = src + (size_t)m * ld + half * 8;
  v8bf lo = *(const v8bf*)(p);
  v8bf hi = *(const v8bf*)(p + 16);
  v16bf r;
#pragma unroll
  for (int i = 0; i < 8; ++i) { r[i] = lo[i]; r[i + 8] = hi[i]; }
  return r;
}

// B fragment (32x16 bf16, K x N) from N-major storage srcT[n*ld + k].
// Lane L: n = L%16, K = half*8 + [0..7] and K = 16 + half*8 + [0..7].
__device__ __forceinline__ v16bf load_frag_b(const __bf16* srcT, int ld) {
  const int lane = threadIdx.x & 31;
  const int n    = lane & 15;
  const int half = lane >> 4;
  const __bf16* p = srcT + (size_t)n * ld + half * 8;
  v8bf lo = *(const v8bf*)(p);
  v8bf hi = *(const v8bf*)(p + 16);
  v16bf r;
#pragma unroll
  for (int i = 0; i < 8; ++i) { r[i] = lo[i]; r[i + 8] = hi[i]; }
  return r;
}

__device__ __forceinline__ v8f wmma_bf16(v16bf a, v16bf b, v8f c) {
  return __builtin_amdgcn_wmma_f32_16x16x32_bf16(
      /*neg_a=*/false, a, /*neg_b=*/false, b,
      /*c_mod=*/(short)0, c, /*reuse_a=*/false, /*reuse_b=*/false);
}

// ---------------------------------------------------------------------------
// K0a: fp32 -> bf16 elementwise (vectorized 4-wide)
// ---------------------------------------------------------------------------
__global__ __launch_bounds__(256) void cvt_bf16_kernel(
    const float* __restrict__ src, __bf16* __restrict__ dst, int n4) {
  int i = blockIdx.x * blockDim.x + threadIdx.x;
  if (i >= n4) return;
  v4f v = ((const v4f*)src)[i];
  v4bf o;
#pragma unroll
  for (int j = 0; j < 4; ++j) o[j] = f2bf(v[j]);
  ((v4bf*)dst)[i] = o;
}

// ---------------------------------------------------------------------------
// K0b: transpose + convert:  src[R][Ccols] fp32  ->  dst[Ccols][R] bf16
// ---------------------------------------------------------------------------
__global__ __launch_bounds__(256) void transpose_cvt_kernel(
    const float* __restrict__ src, __bf16* __restrict__ dst, int R, int Ccols) {
  int i = blockIdx.x * blockDim.x + threadIdx.x;
  if (i >= R * Ccols) return;
  int r = i % R;
  int c = i / R;
  dst[i] = f2bf(src[(size_t)r * Ccols + c]);
}

// ---------------------------------------------------------------------------
// K1: QKV GEMM.  [B*T,384] x [384,1152] + bias -> q (row-major bf16),
//     k (row-major bf16), vT (transposed bf16: vT[b][c][t]).
//     One wave per 16x64 output strip.  sched_barrier(0) pins all 24 b128
//     B-fragment loads of a tile before the 12-WMMA accumulation chain.
// ---------------------------------------------------------------------------
__global__ __launch_bounds__(256) void qkv_gemm_kernel(
    const __bf16* __restrict__ xbf, const __bf16* __restrict__ WqkvT,
    const float* __restrict__ bqkv, __bf16* __restrict__ q,
    __bf16* __restrict__ k, __bf16* __restrict__ vT) {
  const int lane = threadIdx.x & 31;
  const int half = lane >> 4;
  const int nlo  = lane & 15;
  const int wid  = blockIdx.x * (blockDim.x >> 5) + (threadIdx.x >> 5);
  const int NGRP = CH3 / 64;             // 18 n-groups of 64 cols
  const int mt   = wid / NGRP;
  const int ng   = wid % NGRP;
  const int row0 = mt * 16;              // global bt row
  const int n0   = ng * 64;
  const int b    = row0 >> 8;            // /SEQ
  const int t0   = row0 & 255;

  v16bf af[12];
#pragma unroll
  for (int i = 0; i < 12; ++i)
    af[i] = load_frag_a(xbf + (size_t)row0 * CH + i * 32, CH);

#pragma unroll
  for (int ns = 0; ns < 4; ++ns) {
    const int ncol = n0 + ns * 16;
    // bulk-preload all B fragments for this tile; keep them live across the
    // sched barrier so the allocator cannot coalesce them into one buffer.
    v16bf bfr[12];
#pragma unroll
    for (int i = 0; i < 12; ++i)
      bfr[i] = load_frag_b(WqkvT + (size_t)ncol * CH + i * 32, CH);
    __builtin_amdgcn_sched_barrier(0);
    v8f acc = {};
#pragma unroll
    for (int i = 0; i < 12; ++i) acc = wmma_bf16(af[i], bfr[i], acc);

    const int   ncg  = ncol + nlo;
    const float bias = bqkv[ncg];
#pragma unroll
    for (int r = 0; r < 8; ++r) acc[r] += bias;

    if (ncg < CH) {                      // q, row-major
#pragma unroll
      for (int r = 0; r < 8; ++r)
        q[(size_t)(row0 + r + 8 * half) * CH + ncg] = f2bf(acc[r]);
    } else if (ncg < 2 * CH) {           // k, row-major
#pragma unroll
      for (int r = 0; r < 8; ++r)
        k[(size_t)(row0 + r + 8 * half) * CH + (ncg - CH)] = f2bf(acc[r]);
    } else {                             // v, transposed: vT[b][c][t]
      v8bf pk;
#pragma unroll
      for (int r = 0; r < 8; ++r) pk[r] = f2bf(acc[r]);
      *(v8bf*)(vT + (size_t)(b * CH + (ncg - 2 * CH)) * SEQ + t0 + 8 * half) = pk;
    }
  }
}

// ---------------------------------------------------------------------------
// K2: causal flash attention.  One wave per 16 query rows; online softmax
//     over 32-key blocks.  Inner loops use chunked fragment preloads pinned
//     with sched_barrier(0); loads of chunk c+1 may interleave with WMMAs of
//     chunk c (both live in the same inter-barrier region).
// ---------------------------------------------------------------------------
__global__ __launch_bounds__(256) void attention_kernel(
    const __bf16* __restrict__ q, const __bf16* __restrict__ k,
    const __bf16* __restrict__ vT, __bf16* __restrict__ outb) {
  __shared__ __attribute__((aligned(16))) __bf16 pst[8][16][32];  // 8KB
  const int lane = threadIdx.x & 31;
  const int w    = threadIdx.x >> 5;
  const int wid  = blockIdx.x * 8 + w;
  const int b    = wid >> 4;             // SEQ/16 = 16 q-tiles per batch
  const int tq   = (wid & 15) << 4;
  const int half = lane >> 4;
  const int nlo  = lane & 15;
  const float scale = 0.051031036307982884f;  // 1/sqrt(384)

  v16bf qf[12];
#pragma unroll
  for (int i = 0; i < 12; ++i)
    qf[i] = load_frag_a(q + (size_t)(b * SEQ + tq) * CH + i * 32, CH);

  v8f acc[24];
#pragma unroll
  for (int ct = 0; ct < 24; ++ct) acc[ct] = (v8f){};
  float mrow[8], lrow[8];
#pragma unroll
  for (int r = 0; r < 8; ++r) { mrow[r] = -3.0e38f; lrow[r] = 0.0f; }

  const int send = tq + 16;              // causal: keys <= last row of tile
  for (int s0 = 0; s0 < send; s0 += 32) {
    if (s0 + 32 < send)
      __builtin_prefetch(k + (size_t)(b * SEQ + s0 + 32) * CH, 0, 0);

    const __bf16* kbase0 = k + (size_t)(b * SEQ + s0) * CH;
    const __bf16* kbase1 = k + (size_t)(b * SEQ + s0 + 16) * CH;

    // ---- scores: two 16x16 tiles; 3 chunks of 4 K-steps (8 fragments) ----
    v8f st0 = {}, st1 = {};
#pragma unroll
    for (int c = 0; c < 3; ++c) {
      v16bf kb[8];
#pragma unroll
      for (int j = 0; j < 4; ++j) {
        kb[2 * j]     = load_frag_b(kbase0 + (c * 4 + j) * 32, CH);
        kb[2 * j + 1] = load_frag_b(kbase1 + (c * 4 + j) * 32, CH);
      }
      __builtin_amdgcn_sched_barrier(0);
#pragma unroll
      for (int j = 0; j < 4; ++j) {
        st0 = wmma_bf16(qf[c * 4 + j], kb[2 * j],     st0);
        st1 = wmma_bf16(qf[c * 4 + j], kb[2 * j + 1], st1);
      }
    }
    // ---- causal mask + scale (C layout: row = r + 8*half, col = nlo) ----
#pragma unroll
    for (int r = 0; r < 8; ++r) {
      const int tg = tq + r + 8 * half;
      st0[r] = (s0 + nlo      <= tg) ? st0[r] * scale : -3.0e38f;
      st1[r] = (s0 + 16 + nlo <= tg) ? st1[r] * scale : -3.0e38f;
    }
    // ---- online softmax: per-row max/sum via 16-lane xor reductions ----
    float alpha[8];
#pragma unroll
    for (int r = 0; r < 8; ++r) {
      float mx = fmaxf(st0[r], st1[r]);
      mx = fmaxf(mx, __shfl_xor(mx, 1, 32));
      mx = fmaxf(mx, __shfl_xor(mx, 2, 32));
      mx = fmaxf(mx, __shfl_xor(mx, 4, 32));
      mx = fmaxf(mx, __shfl_xor(mx, 8, 32));
      const float mnew = fmaxf(mrow[r], mx);
      alpha[r] = __expf(mrow[r] - mnew);
      const float p0 = __expf(st0[r] - mnew);
      const float p1 = __expf(st1[r] - mnew);
      st0[r] = p0; st1[r] = p1;
      float rs = p0 + p1;
      rs += __shfl_xor(rs, 1, 32);
      rs += __shfl_xor(rs, 2, 32);
      rs += __shfl_xor(rs, 4, 32);
      rs += __shfl_xor(rs, 8, 32);
      lrow[r] = lrow[r] * alpha[r] + rs;
      mrow[r] = mnew;
    }
#pragma unroll
    for (int ct = 0; ct < 24; ++ct)
#pragma unroll
      for (int r = 0; r < 8; ++r) acc[ct][r] *= alpha[r];

    // ---- P: C layout -> A layout through per-wave LDS tile ----
#pragma unroll
    for (int r = 0; r < 8; ++r) {
      pst[w][r + 8 * half][nlo]      = f2bf(st0[r]);
      pst[w][r + 8 * half][16 + nlo] = f2bf(st1[r]);
    }
    asm volatile("s_wait_dscnt 0" ::: "memory");
    v16bf pf = load_frag_a(&pst[w][0][0], 32);

    // ---- O += P @ V : 4 chunks of 6 v-fragments ----
    const __bf16* vbase = vT + (size_t)(b * CH) * SEQ + s0;
#pragma unroll
    for (int c = 0; c < 4; ++c) {
      v16bf vb[6];
#pragma unroll
      for (int j = 0; j < 6; ++j)
        vb[j] = load_frag_b(vbase + (size_t)((c * 6 + j) * 16) * SEQ, SEQ);
      __builtin_amdgcn_sched_barrier(0);
#pragma unroll
      for (int j = 0; j < 6; ++j)
        acc[c * 6 + j] = wmma_bf16(pf, vb[j], acc[c * 6 + j]);
    }
  }

  // ---- normalize + store (bf16 row-major) ----
  float inv[8];
#pragma unroll
  for (int r = 0; r < 8; ++r) inv[r] = 1.0f / lrow[r];
#pragma unroll
  for (int ct = 0; ct < 24; ++ct) {
    const int c = ct * 16 + nlo;
#pragma unroll
    for (int r = 0; r < 8; ++r)
      outb[(size_t)(b * SEQ + tq + r + 8 * half) * CH + c] =
          f2bf(acc[ct][r] * inv[r]);
  }
}

// ---------------------------------------------------------------------------
// K3: projection GEMM.  [B*T,384] x [384,384] + bias -> fp32 output.
// ---------------------------------------------------------------------------
__global__ __launch_bounds__(256) void proj_gemm_kernel(
    const __bf16* __restrict__ outb, const __bf16* __restrict__ WprojT,
    const float* __restrict__ bproj, float* __restrict__ out) {
  const int lane = threadIdx.x & 31;
  const int half = lane >> 4;
  const int nlo  = lane & 15;
  const int wid  = blockIdx.x * (blockDim.x >> 5) + (threadIdx.x >> 5);
  const int NGRP = CH / 64;              // 6
  const int mt   = wid / NGRP;
  const int ng   = wid % NGRP;
  const int row0 = mt * 16;
  const int n0   = ng * 64;

  v16bf af[12];
#pragma unroll
  for (int i = 0; i < 12; ++i)
    af[i] = load_frag_a(outb + (size_t)row0 * CH + i * 32, CH);

#pragma unroll
  for (int ns = 0; ns < 4; ++ns) {
    const int ncol = n0 + ns * 16;
    v16bf bfr[12];
#pragma unroll
    for (int i = 0; i < 12; ++i)
      bfr[i] = load_frag_b(WprojT + (size_t)ncol * CH + i * 32, CH);
    __builtin_amdgcn_sched_barrier(0);
    v8f acc = {};
#pragma unroll
    for (int i = 0; i < 12; ++i) acc = wmma_bf16(af[i], bfr[i], acc);

    const int   ncg  = ncol + nlo;
    const float bias = bproj[ncg];
#pragma unroll
    for (int r = 0; r < 8; ++r)
      out[(size_t)(row0 + r + 8 * half) * CH + ncg] = acc[r] + bias;
  }
}

// ---------------------------------------------------------------------------
// Host-side launcher
// ---------------------------------------------------------------------------
extern "C" void kernel_launch(void* const* d_in, const int* in_sizes, int n_in,
                              void* d_out, int out_size, void* d_ws, size_t ws_size,
                              hipStream_t stream) {
  const float* x     = (const float*)d_in[0];
  const float* Wqkv  = (const float*)d_in[1];
  const float* bqkv  = (const float*)d_in[2];
  const float* Wproj = (const float*)d_in[3];
  const float* bproj = (const float*)d_in[4];
  float* out = (float*)d_out;

  // carve workspace (256B-aligned slices)
  char* p = (char*)d_ws;
  auto carve = [&](size_t bytes) -> char* {
    char* r = p;
    p += (bytes + 255) & ~(size_t)255;
    return r;
  };
  const size_t NTOK = (size_t)BATCH * SEQ;        // 131072
  const size_t NELT = NTOK * CH;                  // 50,331,648
  __bf16* xbf    = (__bf16*)carve(NELT * 2);
  __bf16* qb     = (__bf16*)carve(NELT * 2);
  __bf16* kb     = (__bf16*)carve(NELT * 2);
  __bf16* vTb    = (__bf16*)carve(NELT * 2);
  __bf16* attb   = (__bf16*)carve(NELT * 2);
  __bf16* WqkvT  = (__bf16*)carve((size_t)CH * CH3 * 2);
  __bf16* WprojT = (__bf16*)carve((size_t)CH * CH * 2);

  // K0a: x -> bf16
  {
    int n4 = (int)(NELT / 4);
    cvt_bf16_kernel<<<(n4 + 255) / 256, 256, 0, stream>>>(x, xbf, n4);
  }
  // K0b: transpose weights
  {
    int n = CH * CH3;
    transpose_cvt_kernel<<<(n + 255) / 256, 256, 0, stream>>>(Wqkv, WqkvT, CH, CH3);
  }
  {
    int n = CH * CH;
    transpose_cvt_kernel<<<(n + 255) / 256, 256, 0, stream>>>(Wproj, WprojT, CH, CH);
  }
  // K1: QKV GEMM  (8192 m-tiles * 18 n-groups = 147456 waves, 8 waves/block)
  {
    int waves = (int)(NTOK / 16) * (CH3 / 64);
    qkv_gemm_kernel<<<waves / 8, 256, 0, stream>>>(xbf, WqkvT, bqkv, qb, kb, vTb);
  }
  // K2: attention  (8192 waves, 8 waves/block)
  {
    int waves = (int)(NTOK / 16);
    attention_kernel<<<waves / 8, 256, 0, stream>>>(qb, kb, vTb, attb);
  }
  // K3: projection GEMM  (8192 * 6 = 49152 waves)
  {
    int waves = (int)(NTOK / 16) * (CH / 64);
    proj_gemm_kernel<<<waves / 8, 256, 0, stream>>>(attb, WprojT, bproj, out);
  }
}
